// VoxelNetLiDAREncoder_73813307949096
// MI455X (gfx1250) — compile-verified, matrix operand order checked by reading, not run on hardware
//
#include <hip/hip_runtime.h>

typedef __attribute__((ext_vector_type(16))) __bf16 v16bf;
typedef __attribute__((ext_vector_type(8)))  __bf16 v8bf;
typedef __attribute__((ext_vector_type(8)))  float  v8f;

#define EPSBN 1e-5f

// ---------------------------------------------------------------- utilities
__global__ void k_fill_u32(unsigned* __restrict__ p, unsigned v, long n) {
  long i = (long)blockIdx.x * blockDim.x + threadIdx.x;
  if (i < n) p[i] = v;
}

// y[r,c] = b[c] + sum_k x[r,k]*w[k,c]; accumulate per-channel sum / sumsq
__global__ void k_lin_stats(const float* __restrict__ x, const float* __restrict__ w,
                            const float* __restrict__ b, long R, int K, int N,
                            float* __restrict__ stats) {
  __shared__ float ssum[256];
  __shared__ float ssq[256];
  int tid = threadIdx.x;
  for (int c = tid; c < N; c += blockDim.x) { ssum[c] = 0.f; ssq[c] = 0.f; }
  __syncthreads();
  long i = (long)blockIdx.x * blockDim.x + tid;
  if (i < R * (long)N) {
    long r = i / N;
    int  c = (int)(i % N);
    float acc = b[c];
    for (int k = 0; k < K; ++k) acc += x[r * K + k] * w[(long)k * N + c];
    atomicAdd(&ssum[c], acc);
    atomicAdd(&ssq[c], acc * acc);
  }
  __syncthreads();
  for (int c = tid; c < N; c += blockDim.x) {
    atomicAdd(&stats[c], ssum[c]);
    atomicAdd(&stats[N + c], ssq[c]);
  }
}

// scale = gamma * rsqrt(var+eps), shift = beta - mean*scale
__global__ void k_finalize(const float* __restrict__ stats, const float* __restrict__ gamma,
                           const float* __restrict__ beta, float count, int N,
                           float* __restrict__ scsh) {
  int c = blockIdx.x * blockDim.x + threadIdx.x;
  if (c < N) {
    float mean = stats[c] / count;
    float var  = stats[N + c] / count - mean * mean;
    float sc   = gamma[c] * rsqrtf(var + EPSBN);
    scsh[c]     = sc;
    scsh[N + c] = beta[c] - mean * sc;
  }
}

// ---------------------------------------------------------------- VFE layers
__global__ void k_vfe1_apply(const float* __restrict__ x, const float* __restrict__ w,
                             const float* __restrict__ b, const float* __restrict__ scsh,
                             float* __restrict__ f1, int NVv) {
  int i = blockIdx.x * blockDim.x + threadIdx.x;
  if (i >= NVv * 64) return;
  int v = i >> 6, c = i & 63;
  float sc = scsh[c], sh = scsh[64 + c], bb = b[c];
  float mx = 0.f;  // relu'd values are >= 0
  for (int p = 0; p < 32; ++p) {
    const float* xp = x + ((long)v * 32 + p) * 5;
    float y = bb;
    #pragma unroll
    for (int k = 0; k < 5; ++k) y += xp[k] * w[k * 64 + c];
    y = y * sc + sh;
    mx = fmaxf(mx, y);
  }
  f1[(long)v * 64 + c] = mx;
}

__global__ void k_vfe2_apply(const float* __restrict__ f1, const float* __restrict__ w,
                             const float* __restrict__ b, const float* __restrict__ scsh,
                             __bf16* __restrict__ f2, int NVv) {
  int i = blockIdx.x * blockDim.x + threadIdx.x;
  if (i >= NVv * 128) return;
  int v = i >> 7, c = i & 127;
  float y = b[c];
  const float* xr = f1 + (long)v * 64;
  for (int k = 0; k < 64; ++k) y += xr[k] * w[k * 128 + c];
  y = y * scsh[c] + scsh[128 + c];
  f2[i] = (__bf16)fmaxf(y, 0.f);
}

// ---------------------------------------------------------------- scatter
__global__ void k_winner(const int* __restrict__ coords, int NVv, int* __restrict__ winner) {
  int n = blockIdx.x * blockDim.x + threadIdx.x;
  if (n >= NVv) return;
  int d = coords[n * 3], h = coords[n * 3 + 1], w = coords[n * 3 + 2];
  atomicMax(&winner[(d * 128 + h) * 128 + w], n);
}

__global__ void k_scatter(const int* __restrict__ coords, const int* __restrict__ winner,
                          const __bf16* __restrict__ f2, __bf16* __restrict__ grid0, int NVv) {
  int i = blockIdx.x * blockDim.x + threadIdx.x;
  if (i >= NVv * 128) return;
  int n = i >> 7, c = i & 127;
  int d = coords[n * 3], h = coords[n * 3 + 1], w = coords[n * 3 + 2];
  int cell = (d * 128 + h) * 128 + w;
  if (winner[cell] == n) grid0[(long)cell * 128 + c] = f2[i];
}

// ---------------------------------------------------------------- weight repack
// K-order: k = (kd*9+kh*3+kw)*IC + ic. Packed per-lane WMMA-B fragments:
// pk[(((ks*nN + nt)*32 + lane)*16 + e]; lane<16 holds K {0..7,16..23},
// lane>=16 holds K {8..15,24..31}; N = nt*16 + lane%16.
__global__ void k_repack(const float* __restrict__ w, int IC, int OC, __bf16* __restrict__ pk) {
  int nN = OC >> 4;
  long total = (long)((IC * 27) >> 5) * nN * 32 * 16;
  long i = (long)blockIdx.x * blockDim.x + threadIdx.x;
  if (i >= total) return;
  int e = (int)(i & 15);
  int lane = (int)((i >> 4) & 31);
  long t = i >> 9;
  int nt = (int)(t % nN);
  int ks = (int)(t / nN);
  int n = nt * 16 + (lane & 15);
  int khalf = lane >> 4;
  int koff = (e < 8 ? e : e + 8) + khalf * 8;
  int k = ks * 32 + koff;
  int ksp = k / IC;
  int ic = k - ksp * IC;
  pk[i] = (__bf16)w[((long)n * IC + ic) * 27 + ksp];
}

// ---------------------------------------------------------------- conv (implicit GEMM, WMMA bf16)
// Compile-time dims (all powers of two): index math folds to shifts/masks.
// One wave computes 32 flattened output positions x up to 256 out channels, 128 per pass:
// two A fragments share every B fragment (1.25 loads per WMMA).
// __launch_bounds__(256,1): raise per-wave VGPR budget so the 128 accumulator
// VGPRs + fragments stay in registers (no scratch spills).
template <int ID, int IH, int IW, int IC, int OD, int OH, int OW, int OC>
__global__ __launch_bounds__(256, 1) void k_conv_wmma(
    const __bf16* __restrict__ in, const __bf16* __restrict__ pw,
    const float* __restrict__ bias, float* __restrict__ out) {
  constexpr int nN = OC / 16;        // N tiles (8 or 16)
  constexpr int NG = nN / 8;         // groups of 8 N tiles (1 or 2)
  constexpr int kcSteps = IC / 32;
  constexpr int nPos = OD * OH * OW;
  const int lane = threadIdx.x & 31;
  const int wid = (int)((blockIdx.x * blockDim.x + threadIdx.x) >> 5);
  const int pBase = wid * 32;
  if (pBase >= nPos) return;
  const int m = lane & 15;   // row within A fragment / column within D
  const int hi = lane >> 4;  // which K half-chunks this lane holds

  // per-fragment input coordinates for this lane's row
  int od_[2], ih0_[2], iw0_[2];
  #pragma unroll
  for (int f = 0; f < 2; ++f) {
    int p = pBase + f * 16 + m;
    int ow = p % OW;
    int oh = (p / OW) % OH;
    od_[f] = p / (OW * OH);
    ih0_[f] = 2 * oh - 1;
    iw0_[f] = 2 * ow - 1;
  }

  #pragma unroll 1
  for (int g = 0; g < NG; ++g) {
    v8f acc[2][8];
    #pragma unroll
    for (int f = 0; f < 2; ++f)
      #pragma unroll
      for (int t = 0; t < 8; ++t)
        #pragma unroll
        for (int q = 0; q < 8; ++q) acc[f][t][q] = 0.f;

    #pragma unroll 1
    for (int kd = 0; kd < 3; ++kd) {
      #pragma unroll 1
      for (int kh = 0; kh < 3; ++kh) {
        #pragma unroll 1
        for (int kw = 0; kw < 3; ++kw) {
          const int ksp = (kd * 3 + kh) * 3 + kw;
          long base_[2];
          bool ok_[2];
          #pragma unroll
          for (int f = 0; f < 2; ++f) {
            int id = 2 * od_[f] + kd - 1;
            int ih = ih0_[f] + kh;
            int iw = iw0_[f] + kw;
            ok_[f] = ((unsigned)id < (unsigned)ID) && ((unsigned)ih < (unsigned)IH) &&
                     ((unsigned)iw < (unsigned)IW);
            base_[f] = (((long)id * IH + ih) * IW + iw) * IC + hi * 8;
          }
          const __bf16* pb0 = pw + (((long)ksp * kcSteps * nN + g * 8) * 32 + lane) * 16;
          #pragma unroll 1
          for (int kcs = 0; kcs < kcSteps; ++kcs) {
            // ---- two A fragments: two contiguous b128 loads each (zero at borders)
            v16bf A[2];
            #pragma unroll
            for (int f = 0; f < 2; ++f) {
              v8bf alo, ahi;
              #pragma unroll
              for (int q = 0; q < 8; ++q) { alo[q] = (__bf16)0.f; ahi[q] = (__bf16)0.f; }
              if (ok_[f]) {
                const __bf16* ap = in + base_[f] + kcs * 32;
                alo = *(const v8bf*)(ap);
                ahi = *(const v8bf*)(ap + 16);
              }
              #pragma unroll
              for (int q = 0; q < 8; ++q) { A[f][q] = alo[q]; A[f][q + 8] = ahi[q]; }
            }

            const __bf16* pb = pb0 + (long)kcs * nN * 512;
            __builtin_prefetch(pb + (long)nN * 512, 0, 1);  // next K-step's fragments
            // ---- 8 B fragments, each feeding two WMMAs (B reuse across M)
            #pragma unroll
            for (int t = 0; t < 8; ++t) {
              v8bf blo = *(const v8bf*)(pb + (long)t * 512);
              v8bf bhi = *(const v8bf*)(pb + (long)t * 512 + 8);
              v16bf B;
              #pragma unroll
              for (int q = 0; q < 8; ++q) { B[q] = blo[q]; B[q + 8] = bhi[q]; }
              acc[0][t] = __builtin_amdgcn_wmma_f32_16x16x32_bf16(
                  false, A[0], false, B, (short)0, acc[0][t], false, false);
              acc[1][t] = __builtin_amdgcn_wmma_f32_16x16x32_bf16(
                  false, A[1], false, B, (short)0, acc[1][t], false, false);
            }
          }
        }
      }
    }

    // ---- store D (+bias): VGPR r holds M=r (lanes 0-15) / M=r+8 (lanes 16-31)
    #pragma unroll
    for (int t = 0; t < 8; ++t) {
      int oc = (g * 8 + t) * 16 + m;
      float bv = bias[oc];
      #pragma unroll
      for (int f = 0; f < 2; ++f)
        #pragma unroll
        for (int r = 0; r < 8; ++r) {
          int p = pBase + f * 16 + r + hi * 8;
          out[(long)p * OC + oc] = acc[f][t][r] + bv;
        }
    }
  }
}

// per-channel stats over fp32 NDHWC conv output
__global__ void k_chan_stats(const float* __restrict__ x, long total, int C,
                             float* __restrict__ stats) {
  __shared__ float ssum[256];
  __shared__ float ssq[256];
  int tid = threadIdx.x;
  for (int c = tid; c < C; c += blockDim.x) { ssum[c] = 0.f; ssq[c] = 0.f; }
  __syncthreads();
  long i = (long)blockIdx.x * blockDim.x + tid;
  if (i < total) {
    float v = x[i];
    int c = (int)(i % C);
    atomicAdd(&ssum[c], v);
    atomicAdd(&ssq[c], v * v);
  }
  __syncthreads();
  for (int c = tid; c < C; c += blockDim.x) {
    atomicAdd(&stats[c], ssum[c]);
    atomicAdd(&stats[C + c], ssq[c]);
  }
}

// BN + ReLU -> bf16 (next conv's input, same NDHWC layout)
__global__ void k_bnrelu_bf16(const float* __restrict__ x, const float* __restrict__ scsh,
                              long total, int C, __bf16* __restrict__ y) {
  long i = (long)blockIdx.x * blockDim.x + threadIdx.x;
  if (i >= total) return;
  int c = (int)(i % C);
  float v = x[i] * scsh[c] + scsh[C + c];
  y[i] = (__bf16)fmaxf(v, 0.f);
}

// final: BN + ReLU, NDHWC(2,16,16,256) -> NCDHW fp32 output
__global__ void k_bnrelu_out(const float* __restrict__ x, const float* __restrict__ scsh,
                             float* __restrict__ out) {
  int i = blockIdx.x * blockDim.x + threadIdx.x;
  if (i >= 512 * 256) return;
  int c = i & 255;
  int pos = i >> 8;
  int w = pos & 15, h = (pos >> 4) & 15, d = pos >> 8;
  float v = x[i] * scsh[c] + scsh[256 + c];
  out[((c * 2 + d) * 16 + h) * 16 + w] = fmaxf(v, 0.f);
}

// ---------------------------------------------------------------- launch
extern "C" void kernel_launch(void* const* d_in, const int* in_sizes, int n_in,
                              void* d_out, int out_size, void* d_ws, size_t ws_size,
                              hipStream_t stream) {
  const float* vfeat = (const float*)d_in[0];
  const int*   coords = (const int*)d_in[1];
  const float* w1 = (const float*)d_in[5];
  const float* b1 = (const float*)d_in[6];
  const float* g1 = (const float*)d_in[7];
  const float* be1 = (const float*)d_in[8];
  const float* w2 = (const float*)d_in[9];
  const float* b2 = (const float*)d_in[10];
  const float* g2 = (const float*)d_in[11];
  const float* be2 = (const float*)d_in[12];
  const float* cw1 = (const float*)d_in[13];
  const float* cb1 = (const float*)d_in[14];
  const float* cg1 = (const float*)d_in[15];
  const float* cbe1 = (const float*)d_in[16];
  const float* cw2 = (const float*)d_in[17];
  const float* cb2 = (const float*)d_in[18];
  const float* cg2 = (const float*)d_in[19];
  const float* cbe2 = (const float*)d_in[20];
  const float* cw3 = (const float*)d_in[21];
  const float* cb3 = (const float*)d_in[22];
  const float* cg3 = (const float*)d_in[23];
  const float* cbe3 = (const float*)d_in[24];

  const int NVv = in_sizes[1] / 3;  // 12000
  const long CELLS = 16L * 128 * 128;

  char* p = (char*)d_ws;
  auto carve = [&](size_t bytes) -> char* {
    char* r = p;
    p += (bytes + 255) & ~(size_t)255;
    return r;
  };
  float*  f1     = (float*)carve((size_t)NVv * 64 * 4);
  __bf16* f2     = (__bf16*)carve((size_t)NVv * 128 * 2);
  int*    winner = (int*)carve((size_t)CELLS * 4);
  __bf16* grid0  = (__bf16*)carve((size_t)CELLS * 128 * 2);
  float*  c1raw  = (float*)carve((size_t)32768 * 128 * 4);
  __bf16* act1   = (__bf16*)carve((size_t)32768 * 128 * 2);
  float*  c2raw  = (float*)carve((size_t)4096 * 256 * 4);
  __bf16* act2   = (__bf16*)carve((size_t)4096 * 256 * 2);
  float*  c3raw  = (float*)carve((size_t)512 * 256 * 4);
  __bf16* pw1    = (__bf16*)carve((size_t)108 * 8 * 32 * 16 * 2);
  __bf16* pw2    = (__bf16*)carve((size_t)108 * 16 * 32 * 16 * 2);
  __bf16* pw3    = (__bf16*)carve((size_t)216 * 16 * 32 * 16 * 2);
  float*  stats  = (float*)carve(512 * 4);
  float*  scsh   = (float*)carve(512 * 4);

  auto cdiv = [](long a, long b) -> int { return (int)((a + b - 1) / b); };
  const int T = 256;

  // weight repack (independent of everything else)
  k_repack<<<cdiv(108L * 8 * 32 * 16, T), T, 0, stream>>>(cw1, 128, 128, pw1);
  k_repack<<<cdiv(108L * 16 * 32 * 16, T), T, 0, stream>>>(cw2, 128, 256, pw2);
  k_repack<<<cdiv(216L * 16 * 32 * 16, T), T, 0, stream>>>(cw3, 256, 256, pw3);

  // ---- VFE1
  k_fill_u32<<<cdiv(512, T), T, 0, stream>>>((unsigned*)stats, 0u, 512);
  k_lin_stats<<<cdiv((long)NVv * 32 * 64, T), T, 0, stream>>>(vfeat, w1, b1, (long)NVv * 32, 5, 64, stats);
  k_finalize<<<1, T, 0, stream>>>(stats, g1, be1, (float)(NVv * 32), 64, scsh);
  k_vfe1_apply<<<cdiv((long)NVv * 64, T), T, 0, stream>>>(vfeat, w1, b1, scsh, f1, NVv);

  // ---- VFE2
  k_fill_u32<<<cdiv(512, T), T, 0, stream>>>((unsigned*)stats, 0u, 512);
  k_lin_stats<<<cdiv((long)NVv * 128, T), T, 0, stream>>>(f1, w2, b2, (long)NVv, 64, 128, stats);
  k_finalize<<<1, T, 0, stream>>>(stats, g2, be2, (float)NVv, 128, scsh);
  k_vfe2_apply<<<cdiv((long)NVv * 128, T), T, 0, stream>>>(f1, w2, b2, scsh, f2, NVv);

  // ---- scatter to dense bf16 NDHWC grid (deterministic last-write-wins)
  k_fill_u32<<<cdiv(CELLS, T), T, 0, stream>>>((unsigned*)winner, 0xFFFFFFFFu, CELLS);
  k_fill_u32<<<cdiv(CELLS * 128 / 2, T), T, 0, stream>>>((unsigned*)grid0, 0u, CELLS * 128 / 2);
  k_winner<<<cdiv(NVv, T), T, 0, stream>>>(coords, NVv, winner);
  k_scatter<<<cdiv((long)NVv * 128, T), T, 0, stream>>>(coords, winner, f2, grid0, NVv);

  // ---- conv1: (16,128,128,128) -> (8,64,64,128)
  {
    long waves = 32768 / 32;
    k_conv_wmma<16, 128, 128, 128, 8, 64, 64, 128>
        <<<cdiv(waves * 32, T), T, 0, stream>>>(grid0, pw1, cb1, c1raw);
    long tot = 32768L * 128;
    k_fill_u32<<<cdiv(512, T), T, 0, stream>>>((unsigned*)stats, 0u, 512);
    k_chan_stats<<<cdiv(tot, T), T, 0, stream>>>(c1raw, tot, 128, stats);
    k_finalize<<<1, T, 0, stream>>>(stats, cg1, cbe1, 32768.f, 128, scsh);
    k_bnrelu_bf16<<<cdiv(tot, T), T, 0, stream>>>(c1raw, scsh, tot, 128, act1);
  }

  // ---- conv2: (8,64,64,128) -> (4,32,32,256)
  {
    long waves = 4096 / 32;
    k_conv_wmma<8, 64, 64, 128, 4, 32, 32, 256>
        <<<cdiv(waves * 32, T), T, 0, stream>>>(act1, pw2, cb2, c2raw);
    long tot = 4096L * 256;
    k_fill_u32<<<cdiv(512, T), T, 0, stream>>>((unsigned*)stats, 0u, 512);
    k_chan_stats<<<cdiv(tot, T), T, 0, stream>>>(c2raw, tot, 256, stats);
    k_finalize<<<1, T, 0, stream>>>(stats, cg2, cbe2, 4096.f, 256, scsh);
    k_bnrelu_bf16<<<cdiv(tot, T), T, 0, stream>>>(c2raw, scsh, tot, 256, act2);
  }

  // ---- conv3: (4,32,32,256) -> (2,16,16,256), output NCDHW fp32
  {
    long waves = 512 / 32;
    k_conv_wmma<4, 32, 32, 256, 2, 16, 16, 256>
        <<<cdiv(waves * 32, T), T, 0, stream>>>(act2, pw3, cb3, c3raw);
    long tot = 512L * 256;
    k_fill_u32<<<cdiv(512, T), T, 0, stream>>>((unsigned*)stats, 0u, 512);
    k_chan_stats<<<cdiv(tot, T), T, 0, stream>>>(c3raw, tot, 256, stats);
    k_finalize<<<1, T, 0, stream>>>(stats, cg3, cbe3, 512.f, 256, scsh);
    k_bnrelu_out<<<cdiv(tot, T), T, 0, stream>>>(c3raw, scsh, (float*)d_out);
  }
}
